// InvariantPointAttention_28621662060851
// MI455X (gfx1250) — matrix operand order repaired and checked by reading, not compiled, not run on hardware
//
#include <hip/hip_runtime.h>
#include <math.h>

// ---------------------------------------------------------------------------
// CDNA5 (gfx1250) wave32 WMMA implementation of Invariant Point Attention.
// All contractions use v_wmma_f32_16x16x32_bf16; z streams get
// global_prefetch_b8; softmax uses full-wave32 shuffle reduction.
// ---------------------------------------------------------------------------

typedef __attribute__((ext_vector_type(16))) __bf16 bf16x16;
typedef __attribute__((ext_vector_type(16))) short  s16x16;
typedef __attribute__((ext_vector_type(8)))  float  f32x8;

#define DEV __device__ __forceinline__

#define NRES   768
#define CS     384
#define CZ     128
#define CH     16
#define HEADS  12
#define PQK    4
#define PV     8
#define PROJ_N 1152          // 192 + 144 + 384 + 432
#define CATK   2112          // H*(CZ + CH + PV*4)
#define INF_F  100000.0f

// float -> bf16 (round to nearest even), bit-level
DEV unsigned short f2bf(float f) {
  union { float f; unsigned int u; } v; v.f = f;
  unsigned int u = v.u;
  u += 0x7FFFu + ((u >> 16) & 1u);
  return (unsigned short)(u >> 16);
}

// A-matrix 16x32 bf16 element->K map (ISA 7.12.2): e<8 -> lhalf*8+e ; e>=8 -> 16+lhalf*8+(e-8)
DEV int kmapA(int e, int lhalf) {
  return (e < 8) ? (lhalf * 8 + e) : (16 + lhalf * 8 + (e - 8));
}

// Load A fragment (16x32, row-major bf16 storage, stride in elements)
DEV bf16x16 load_a_frag(const unsigned short* base, int stride, int m0, int k0, int lane) {
  int lrow = lane & 15, lhalf = lane >> 4;
  const unsigned short* p = base + (size_t)(m0 + lrow) * stride + k0 + lhalf * 8;
  s16x16 t;
#pragma unroll
  for (int e = 0; e < 8; ++e) t[e] = (short)p[e];
#pragma unroll
  for (int e = 0; e < 8; ++e) t[8 + e] = (short)p[16 + e];
  return __builtin_bit_cast(bf16x16, t);
}

// Load B fragment (32x16) from "BT" storage: BT[n][k] row-major (row = output col).
// Lane layout: e<8 -> K=lhalf*8+e, e>=8 -> K=16+lhalf*8+(e-8), col = lane%16.
DEV bf16x16 load_bt_frag(const unsigned short* base, int stride, int n0, int k0, int lane) {
  int lrow = lane & 15, lhalf = lane >> 4;
  const unsigned short* p = base + (size_t)(n0 + lrow) * stride + k0 + lhalf * 8;
  s16x16 t;
#pragma unroll
  for (int e = 0; e < 8; ++e) t[e] = (short)p[e];
#pragma unroll
  for (int e = 0; e < 8; ++e) t[8 + e] = (short)p[16 + e];
  return __builtin_bit_cast(bf16x16, t);
}

DEV f32x8 wmma_bf16(bf16x16 a, bf16x16 b, f32x8 c) {
  return __builtin_amdgcn_wmma_f32_16x16x32_bf16(false, a, false, b, (short)0, c, false, false);
}

// ---------------------------------------------------------------------------
// Small prep kernels
// ---------------------------------------------------------------------------

__global__ void cvt_bf16_kernel(const float* __restrict__ src, unsigned short* __restrict__ dst, int count) {
  int i = blockIdx.x * blockDim.x + threadIdx.x;
  if (i < count) dst[i] = f2bf(src[i]);
}

// Wb (12x128) -> padded 16x128 bf16 BT
__global__ void wbpad_kernel(const float* __restrict__ Wb, unsigned short* __restrict__ wbbt) {
  int i = blockIdx.x * blockDim.x + threadIdx.x;
  if (i >= 16 * CZ) return;
  int hh = i / CZ;
  wbbt[i] = (hh < HEADS) ? f2bf(Wb[i]) : (unsigned short)0;
}

// softplus(head_weights) * sqrt(1/(3 * PQK * 9/2))
__global__ void hw_kernel(const float* __restrict__ w, float* __restrict__ hwbuf) {
  int i = threadIdx.x;
  if (i < HEADS) hwbuf[i] = log1pf(__expf(w[i])) * 0.1360827635f;
}

// ---------------------------------------------------------------------------
// Generic bf16 GEMM: C[M x Nc] = A[M x K] * BT[Nc x K]^T + bias
// grid.x = Nc/64, grid.y = M/16, one wave per 16x64 tile.
// ---------------------------------------------------------------------------
__global__ __launch_bounds__(32)
void gemm_bf16_kernel(const unsigned short* __restrict__ A,
                      const unsigned short* __restrict__ BT,
                      const float* __restrict__ bias,
                      float* __restrict__ C, int K, int Nc) {
  int lane = threadIdx.x;
  int lrow = lane & 15, lhalf = lane >> 4;
  int m0 = blockIdx.y * 16;
  int n0 = blockIdx.x * 64;
  f32x8 acc[4] = {};
  for (int k0 = 0; k0 < K; k0 += 32) {
    bf16x16 a = load_a_frag(A, K, m0, k0, lane);
#pragma unroll
    for (int t = 0; t < 4; ++t) {
      bf16x16 b = load_bt_frag(BT, K, n0 + t * 16, k0, lane);
      acc[t] = wmma_bf16(a, b, acc[t]);
    }
  }
#pragma unroll
  for (int t = 0; t < 4; ++t) {
    int nn = n0 + t * 16 + lrow;
    float bv = bias[nn];
#pragma unroll
    for (int r = 0; r < 8; ++r) {
      int mm = m0 + r + 8 * lhalf;
      C[(size_t)mm * Nc + nn] = acc[t][r] + bv;
    }
  }
}

// ---------------------------------------------------------------------------
// Split projections; build q/k (bf16, K padded to 32), v^T, rotated points,
// rotated v-points transposed (pt-major, padded to 32 rows/head).
// One thread per (n, h).
// ---------------------------------------------------------------------------
__global__ void split_points_kernel(const float* __restrict__ proj,
                                    const float* __restrict__ rot,
                                    const float* __restrict__ trans,
                                    unsigned short* __restrict__ qb,
                                    unsigned short* __restrict__ kb,
                                    unsigned short* __restrict__ vbt,
                                    unsigned short* __restrict__ vptst,
                                    float* __restrict__ qpts,
                                    float* __restrict__ kpts) {
  int idx = blockIdx.x * blockDim.x + threadIdx.x;
  if (idx >= NRES * HEADS) return;
  int n = idx / HEADS, h = idx % HEADS;
  const float* pr = proj + (size_t)n * PROJ_N;
  unsigned short* qrow = qb + ((size_t)h * NRES + n) * 32;
  unsigned short* krow = kb + ((size_t)h * NRES + n) * 32;
#pragma unroll
  for (int c = 0; c < 16; ++c) {
    qrow[c] = f2bf(pr[h * 16 + c]);       qrow[16 + c] = 0;
    krow[c] = f2bf(pr[336 + h * 32 + c]); krow[16 + c] = 0;
    vbt[((size_t)h * 16 + c) * NRES + n] = f2bf(pr[336 + h * 32 + 16 + c]);
  }
  const float* R = rot + (size_t)n * 9;
  const float* T = trans + (size_t)n * 3;
#pragma unroll
  for (int p = 0; p < PQK; ++p) {
    float x = pr[192 + (0 * HEADS + h) * PQK + p];
    float y = pr[192 + (1 * HEADS + h) * PQK + p];
    float zc = pr[192 + (2 * HEADS + h) * PQK + p];
#pragma unroll
    for (int i = 0; i < 3; ++i)
      qpts[((size_t)(n * HEADS + h) * PQK + p) * 3 + i] =
          R[i * 3 + 0] * x + R[i * 3 + 1] * y + R[i * 3 + 2] * zc + T[i];
  }
#pragma unroll
  for (int p = 0; p < PQK + PV; ++p) {
    float x = pr[720 + (0 * HEADS + h) * 12 + p];
    float y = pr[720 + (1 * HEADS + h) * 12 + p];
    float zc = pr[720 + (2 * HEADS + h) * 12 + p];
    float r0 = R[0] * x + R[1] * y + R[2] * zc + T[0];
    float r1 = R[3] * x + R[4] * y + R[5] * zc + T[1];
    float r2 = R[6] * x + R[7] * y + R[8] * zc + T[2];
    if (p < PQK) {
      float* kp = kpts + ((size_t)(n * HEADS + h) * PQK + p) * 3;
      kp[0] = r0; kp[1] = r1; kp[2] = r2;
    } else {
      int pv = p - PQK;
      vptst[((size_t)h * 32 + pv * 3 + 0) * NRES + n] = f2bf(r0);
      vptst[((size_t)h * 32 + pv * 3 + 1) * NRES + n] = f2bf(r1);
      vptst[((size_t)h * 32 + pv * 3 + 2) * NRES + n] = f2bf(r2);
    }
  }
#pragma unroll
  for (int rpad = 24; rpad < 32; ++rpad)
    vptst[((size_t)h * 32 + rpad) * NRES + n] = 0;
}

// ---------------------------------------------------------------------------
// b_pair = z @ Wb^T + bb  as WMMA:
// A = 16 flat (n,m) rows x 128 of z (f32 -> bf16 in regs); B = padded Wb^T.
// 4 row-tiles per wave; global_prefetch_b8 on the next tile's rows.
// ---------------------------------------------------------------------------
__global__ __launch_bounds__(32)
void bpair_kernel(const float* __restrict__ z,
                  const unsigned short* __restrict__ wbbt,
                  const float* __restrict__ bb,
                  float* __restrict__ bpair) {
  int lane = threadIdx.x;
  int lrow = lane & 15, lhalf = lane >> 4;
  size_t base_row = (size_t)blockIdx.x * 64;   // flat row = n*NRES + m
  float bias = (lrow < HEADS) ? bb[lrow] : 0.f;
  for (int tile = 0; tile < 4; ++tile) {
    size_t row0 = base_row + (size_t)tile * 16;
    // prefetch next tile's row for this lane toward L2 while computing
    if (tile < 3)
      __builtin_prefetch(z + (row0 + 16 + lrow) * CZ, 0, 1);
    f32x8 acc = {};
#pragma unroll
    for (int k0 = 0; k0 < CZ; k0 += 32) {
      const float* p = z + (row0 + lrow) * CZ + k0 + lhalf * 8;
      s16x16 t;
#pragma unroll
      for (int e = 0; e < 8; ++e) t[e] = (short)f2bf(p[e]);
#pragma unroll
      for (int e = 0; e < 8; ++e) t[8 + e] = (short)f2bf(p[16 + e]);
      bf16x16 a = __builtin_bit_cast(bf16x16, t);
      bf16x16 b = load_bt_frag(wbbt, CZ, 0, k0, lane);
      acc = wmma_bf16(a, b, acc);
    }
    if (lrow < HEADS) {
#pragma unroll
      for (int r = 0; r < 8; ++r) {
        size_t flat = row0 + r + 8 * lhalf;
        bpair[(size_t)lrow * NRES * NRES + flat] = acc[r] + bias;
      }
    }
  }
}

// ---------------------------------------------------------------------------
// Attention core per (head, 16-row tile): QK WMMA + bias/pt-att/mask,
// full-wave32 softmax over 768 cols (logit strip in LDS), then A*V and
// A*Vpts WMMAs.
// ---------------------------------------------------------------------------
__global__ __launch_bounds__(32)
void attn_kernel(const unsigned short* __restrict__ qb,
                 const unsigned short* __restrict__ kb,
                 const float* __restrict__ bpair,
                 const float* __restrict__ qpts,
                 const float* __restrict__ kpts,
                 const float* __restrict__ hwbuf,
                 const float* __restrict__ mask,
                 const unsigned short* __restrict__ vbt,
                 const unsigned short* __restrict__ vptst,
                 unsigned short* __restrict__ abuf,
                 float* __restrict__ obuf,
                 float* __restrict__ optbuf) {
  __shared__ float slog[16 * NRES];   // 48 KB logit/prob strip
  __shared__ float sqp[16 * 12];      // q_pts for 16 rows

  int lane = threadIdx.x;
  int lrow = lane & 15, lhalf = lane >> 4;
  int h = blockIdx.y;
  int n0 = blockIdx.x * 16;

  if (lane < 16) {
    int n = n0 + lane;
#pragma unroll
    for (int t = 0; t < 12; ++t)
      sqp[lane * 12 + t] = qpts[(size_t)(n * HEADS + h) * 12 + t];
  }
  __syncthreads();

  const float QK_SCALE = 0.1443375673f;  // sqrt(1/48)
  const float B_SCALE  = 0.5773502692f;  // sqrt(1/3)
  float hwv = hwbuf[h];

  bf16x16 aq = load_a_frag(qb + (size_t)h * NRES * 32, 32, n0, 0, lane);
  float maskn[8];
#pragma unroll
  for (int r = 0; r < 8; ++r) maskn[r] = mask[n0 + r + 8 * lhalf];

  for (int mt = 0; mt < NRES / 16; ++mt) {
    int m0 = mt * 16;
    bf16x16 bk = load_bt_frag(kb + (size_t)h * NRES * 32, 32, m0, 0, lane);
    f32x8 c = {};
    c = wmma_bf16(aq, bk, c);
    int m = m0 + lrow;
    float kp[12];
#pragma unroll
    for (int t = 0; t < 12; ++t) kp[t] = kpts[(size_t)(m * HEADS + h) * 12 + t];
    float maskm = mask[m];
#pragma unroll
    for (int r = 0; r < 8; ++r) {
      int nl = r + 8 * lhalf;
      float val = c[r] * QK_SCALE;
      val += B_SCALE * bpair[(size_t)h * NRES * NRES + (size_t)(n0 + nl) * NRES + m];
      float s2 = 0.f;
#pragma unroll
      for (int t = 0; t < 12; ++t) { float d = sqp[nl * 12 + t] - kp[t]; s2 += d * d; }
      val -= 0.5f * hwv * s2;
      val += INF_F * (maskn[r] * maskm - 1.0f);
      slog[nl * NRES + m] = val;
    }
  }
  __syncthreads();

  // Full-wave softmax: each lane owns half of one row; combine via lane^16.
  {
    int row = lrow;
    float* rp = slog + row * NRES + lhalf * (NRES / 2);
    float mx = -3.0e38f;
    for (int m = 0; m < NRES / 2; ++m) mx = fmaxf(mx, rp[m]);
    mx = fmaxf(mx, __shfl_xor(mx, 16, 32));
    float ssum = 0.f;
    for (int m = 0; m < NRES / 2; ++m) { float e = __expf(rp[m] - mx); rp[m] = e; ssum += e; }
    ssum += __shfl_xor(ssum, 16, 32);
    float inv = 1.0f / ssum;
    unsigned short* arow =
        abuf + (size_t)(h * NRES + n0 + row) * NRES + lhalf * (NRES / 2);
    for (int m = 0; m < NRES / 2; ++m) { float p = rp[m] * inv; rp[m] = p; arow[m] = f2bf(p); }
  }
  __syncthreads();

  f32x8 co = {}, cp0 = {}, cp1 = {};
  for (int m0 = 0; m0 < NRES; m0 += 32) {
    s16x16 t;
#pragma unroll
    for (int e = 0; e < 16; ++e)
      t[e] = (short)f2bf(slog[lrow * NRES + m0 + kmapA(e, lhalf)]);
    bf16x16 ap = __builtin_bit_cast(bf16x16, t);
    bf16x16 bv  = load_bt_frag(vbt   + (size_t)h * 16 * NRES, NRES, 0,  m0, lane);
    bf16x16 bp0 = load_bt_frag(vptst + (size_t)h * 32 * NRES, NRES, 0,  m0, lane);
    bf16x16 bp1 = load_bt_frag(vptst + (size_t)h * 32 * NRES, NRES, 16, m0, lane);
    co  = wmma_bf16(ap, bv,  co);
    cp0 = wmma_bf16(ap, bp0, cp0);
    cp1 = wmma_bf16(ap, bp1, cp1);
  }
#pragma unroll
  for (int r = 0; r < 8; ++r) {
    int n = n0 + r + 8 * lhalf;
    obuf[(size_t)n * (HEADS * CH) + h * CH + lrow] = co[r];
    optbuf[((size_t)n * HEADS + h) * 24 + lrow] = cp0[r];
    if (lrow < 8) optbuf[((size_t)n * HEADS + h) * 24 + 16 + lrow] = cp1[r];
  }
}

// ---------------------------------------------------------------------------
// o_pair[n] = a[:, n, :] @ z[n]  (12-pad-16 x 768 x 128 GEMM per residue).
// Streams z a second time, f32->bf16 in regs, prefetching the next K-tile.
// ---------------------------------------------------------------------------
__global__ __launch_bounds__(32)
void opair_kernel(const unsigned short* __restrict__ abuf,
                  const float* __restrict__ z,
                  float* __restrict__ opairbuf) {
  int lane = threadIdx.x;
  int lrow = lane & 15, lhalf = lane >> 4;
  int n = blockIdx.x;
  f32x8 acc[8] = {};
  for (int m0 = 0; m0 < NRES; m0 += 32) {
    // prefetch the next 32-row z block (1 row per lane, 2 prefetches per row, covers all 32 rows)
    if (m0 + 32 < NRES) {
      __builtin_prefetch(z + ((size_t)n * NRES + m0 + 32 + lane) * CZ, 0, 1);
      __builtin_prefetch(z + ((size_t)n * NRES + m0 + 32 + lane) * CZ + 64, 0, 1);
    }
    s16x16 ta;
    if (lrow < HEADS) {
      const unsigned short* p = abuf + ((size_t)lrow * NRES + n) * NRES + m0 + lhalf * 8;
#pragma unroll
      for (int e = 0; e < 8; ++e) ta[e] = (short)p[e];
#pragma unroll
      for (int e = 0; e < 8; ++e) ta[8 + e] = (short)p[16 + e];
    } else {
#pragma unroll
      for (int e = 0; e < 16; ++e) ta[e] = 0;
    }
    bf16x16 a = __builtin_bit_cast(bf16x16, ta);
    const float* zp = z + ((size_t)n * NRES + m0 + lhalf * 8) * CZ + lrow;
#pragma unroll
    for (int ct = 0; ct < 8; ++ct) {
      s16x16 tb;
#pragma unroll
      for (int e = 0; e < 16; ++e)
        tb[e] = (short)f2bf(zp[(size_t)(e < 8 ? e : e + 8) * CZ + ct * 16]);
      acc[ct] = wmma_bf16(a, __builtin_bit_cast(bf16x16, tb), acc[ct]);
    }
  }
#pragma unroll
  for (int ct = 0; ct < 8; ++ct) {
#pragma unroll
    for (int r = 0; r < 8; ++r) {
      int hh = r + 8 * lhalf;
      if (hh < HEADS)
        opairbuf[(size_t)n * (HEADS * CZ) + hh * CZ + ct * 16 + lrow] = acc[ct][r];
    }
  }
}

// ---------------------------------------------------------------------------
// Build concat row (inverse-rotate o_pt, norms) directly in bf16.
// ---------------------------------------------------------------------------
__global__ void cat_kernel(const float* __restrict__ obuf,
                           const float* __restrict__ optbuf,
                           const float* __restrict__ opairbuf,
                           const float* __restrict__ rot,
                           const float* __restrict__ trans,
                           unsigned short* __restrict__ catb) {
  int idx = blockIdx.x * blockDim.x + threadIdx.x;
  if (idx >= NRES * CATK) return;
  int n = idx / CATK, j = idx % CATK;
  float val;
  if (j < 192) {
    val = obuf[(size_t)n * 192 + j];
  } else if (j < 576) {
    int jj = j - 192;
    bool isnorm = jj >= 288;
    int axis = isnorm ? 0 : jj / 96;
    int hp   = isnorm ? (jj - 288) : (jj % 96);
    int hh = hp / PV, p = hp % PV;
    const float* g = optbuf + ((size_t)n * HEADS + hh) * 24 + p * 3;
    const float* R = rot + (size_t)n * 9;
    const float* T = trans + (size_t)n * 3;
    float t0 = g[0] - T[0], t1 = g[1] - T[1], t2 = g[2] - T[2];
    if (isnorm) {
      float l0 = R[0] * t0 + R[3] * t1 + R[6] * t2;
      float l1 = R[1] * t0 + R[4] * t1 + R[7] * t2;
      float l2 = R[2] * t0 + R[5] * t1 + R[8] * t2;
      val = sqrtf(l0 * l0 + l1 * l1 + l2 * l2 + 1e-8f);
    } else {
      val = R[axis] * t0 + R[3 + axis] * t1 + R[6 + axis] * t2;
    }
  } else {
    val = opairbuf[(size_t)n * (HEADS * CZ) + (j - 576)];
  }
  catb[idx] = f2bf(val);
}

// ---------------------------------------------------------------------------
// Host launch
// ---------------------------------------------------------------------------
static inline size_t align256(size_t x) { return (x + 255) & ~(size_t)255; }

extern "C" void kernel_launch(void* const* d_in, const int* in_sizes, int n_in,
                              void* d_out, int out_size, void* d_ws, size_t ws_size,
                              hipStream_t stream) {
  (void)in_sizes; (void)n_in; (void)out_size; (void)ws_size;

  const float* s     = (const float*)d_in[0];
  const float* z     = (const float*)d_in[1];
  const float* mask  = (const float*)d_in[2];
  const float* rot   = (const float*)d_in[3];
  const float* trans = (const float*)d_in[4];
  const float* Wq    = (const float*)d_in[5];
  const float* bq    = (const float*)d_in[6];
  const float* Wqp   = (const float*)d_in[7];
  const float* bqp   = (const float*)d_in[8];
  const float* Wkv   = (const float*)d_in[9];
  const float* bkv   = (const float*)d_in[10];
  const float* Wkvp  = (const float*)d_in[11];
  const float* bkvp  = (const float*)d_in[12];
  const float* Wb    = (const float*)d_in[13];
  const float* bb    = (const float*)d_in[14];
  const float* hwts  = (const float*)d_in[15];
  const float* Wout  = (const float*)d_in[16];
  const float* bout  = (const float*)d_in[17];
  float* out = (float*)d_out;

  char* ws = (char*)d_ws;
  size_t off = 0;
  auto carve = [&](size_t bytes) -> void* {
    void* p = ws + off; off = align256(off + bytes); return p;
  };

  unsigned short* sb      = (unsigned short*)carve((size_t)NRES * CS * 2);
  unsigned short* wallb   = (unsigned short*)carve((size_t)PROJ_N * CS * 2);
  float*          biasall = (float*)carve((size_t)PROJ_N * 4);
  unsigned short* woutb   = (unsigned short*)carve((size_t)CS * CATK * 2);
  unsigned short* wbbt    = (unsigned short*)carve((size_t)16 * CZ * 2);
  float*          hwbuf   = (float*)carve(64);
  float*          proj    = (float*)carve((size_t)NRES * PROJ_N * 4);
  unsigned short* qb      = (unsigned short*)carve((size_t)HEADS * NRES * 32 * 2);
  unsigned short* kb      = (unsigned short*)carve((size_t)HEADS * NRES * 32 * 2);
  unsigned short* vbt     = (unsigned short*)carve((size_t)HEADS * 16 * NRES * 2);
  unsigned short* vptst   = (unsigned short*)carve((size_t)HEADS * 32 * NRES * 2);
  float*          qpts    = (float*)carve((size_t)NRES * HEADS * PQK * 3 * 4);
  float*          kpts    = (float*)carve((size_t)NRES * HEADS * PQK * 3 * 4);
  float*          bpair   = (float*)carve((size_t)HEADS * NRES * NRES * 4);
  unsigned short* abuf    = (unsigned short*)carve((size_t)HEADS * NRES * NRES * 2);
  float*          obuf    = (float*)carve((size_t)NRES * HEADS * CH * 4);
  float*          optbuf  = (float*)carve((size_t)NRES * HEADS * 24 * 4);
  float*          opairbuf= (float*)carve((size_t)NRES * HEADS * CZ * 4);
  unsigned short* catb    = (unsigned short*)carve((size_t)NRES * CATK * 2);

  auto cvt = [&](const float* src, unsigned short* dst, int count) {
    cvt_bf16_kernel<<<(count + 255) / 256, 256, 0, stream>>>(src, dst, count);
  };

  // ---- precision packing ----
  cvt(s, sb, NRES * CS);
  cvt(Wq,   wallb,              192 * CS);
  cvt(Wqp,  wallb + 192 * CS,   144 * CS);
  cvt(Wkv,  wallb + 336 * CS,   384 * CS);
  cvt(Wkvp, wallb + 720 * CS,   432 * CS);
  cvt(Wout, woutb, CS * CATK);
  hipMemcpyAsync(biasall,        bq,   192 * 4, hipMemcpyDeviceToDevice, stream);
  hipMemcpyAsync(biasall + 192,  bqp,  144 * 4, hipMemcpyDeviceToDevice, stream);
  hipMemcpyAsync(biasall + 336,  bkv,  384 * 4, hipMemcpyDeviceToDevice, stream);
  hipMemcpyAsync(biasall + 720,  bkvp, 432 * 4, hipMemcpyDeviceToDevice, stream);
  wbpad_kernel<<<(16 * CZ + 255) / 256, 256, 0, stream>>>(Wb, wbbt);
  hw_kernel<<<1, 32, 0, stream>>>(hwts, hwbuf);

  // ---- fused projection GEMM: proj = s @ [Wq;Wqp;Wkv;Wkvp]^T + bias ----
  gemm_bf16_kernel<<<dim3(PROJ_N / 64, NRES / 16), 32, 0, stream>>>(
      sb, wallb, biasall, proj, CS, PROJ_N);

  // ---- split heads, rotate points ----
  split_points_kernel<<<(NRES * HEADS + 255) / 256, 256, 0, stream>>>(
      proj, rot, trans, qb, kb, vbt, vptst, qpts, kpts);

  // ---- pair bias: b_pair = z @ Wb^T + bb (4 tiles/wave + prefetch) ----
  bpair_kernel<<<(NRES * NRES) / 64, 32, 0, stream>>>(z, wbbt, bb, bpair);

  // ---- attention core (per head, per 16-row tile) ----
  attn_kernel<<<dim3(NRES / 16, HEADS), 32, 0, stream>>>(
      qb, kb, bpair, qpts, kpts, hwbuf, mask, vbt, vptst, abuf, obuf, optbuf);

  // ---- o_pair per residue ----
  opair_kernel<<<NRES, 32, 0, stream>>>(abuf, z, opairbuf);

  // ---- concat (with inverse rotation + norms) ----
  cat_kernel<<<(NRES * CATK + 255) / 256, 256, 0, stream>>>(
      obuf, optbuf, opairbuf, rot, trans, catb);

  // ---- output GEMM: out = cat @ Wout^T + bout ----
  gemm_bf16_kernel<<<dim3(CS / 64, NRES / 16), 32, 0, stream>>>(
      catb, woutb, bout, out, CATK, CS);
}
